// TrajectoryPredictor_1486058684739
// MI455X (gfx1250) — compile-verified, matrix operand order checked by reading, not run on hardware
//
#include <hip/hip_runtime.h>
#include <hip/hip_bf16.h>

// ---------------- problem constants ----------------
constexpr int Bc = 128, Nn = 256, Tt = 50, Hh = 64, Ee = 64;
constexpr int BN = Bc * Nn;
constexpr float THR2 = 4.0f;           // THR^2
constexpr int FUT = 30;

typedef __attribute__((ext_vector_type(16))) _Float16 v16h;
typedef __attribute__((ext_vector_type(8)))  float    v8f;

// K-offset inside a 16x16x32 f16 A/B fragment for half-index j (0..15),
// per CDNA5 ISA 7.12.2 ("16-bit A-Matrix 16x32"): VGPR0..3 hold K={0..7},
// VGPR4..7 hold K={16..23} for lanes 0-15; upper half-wave adds +8.
__device__ __forceinline__ int AK(int j, int half) {
  int vg = j >> 1, s = j & 1;
  int base = (vg < 4) ? (vg << 1) : (16 + ((vg - 4) << 1));
  return base + s + (half << 3);
}

__device__ __forceinline__ v8f wmma16(v16h a, v16h b, v8f c) {
  // D = A(16x32 f16) * B(32x16 f16) + C(f32)
  return __builtin_amdgcn_wmma_f32_16x16x32_f16(
      /*neg_a=*/false, a, /*neg_b=*/false, b,
      /*c_mod=*/(short)0, c, /*reuse_a=*/false, /*reuse_b=*/false);
}

__device__ __forceinline__ void lds_fence() {
  __builtin_amdgcn_wave_barrier();
  asm volatile("s_wait_dscnt 0x0" ::: "memory");
}

// gfx1250 has native v_tanh_f32; use it when the builtin exists.
#if __has_builtin(__builtin_amdgcn_tanhf)
__device__ __forceinline__ float tanh_(float x) { return __builtin_amdgcn_tanhf(x); }
#else
__device__ __forceinline__ float tanh_(float x) { return tanhf(x); }
#endif
__device__ __forceinline__ float sigmoid_(float x) {
  return 0.5f * tanh_(0.5f * x) + 0.5f;   // 1 TRANS op on gfx1250
}

// =====================================================================
// Kernel 1: LSTM encoder.  grid = BN/(16*8) = 256 blocks, 256 threads.
// Each wave owns 16 sequences; per step: G[16x256] = H[16x64] @ W_hh^T
// via 32 x v_wmma_f32_16x16x32_f16.
// W_hh and the recurrent h state are kept in *fragment-linear* LDS
// order, so each WMMA operand is two contiguous ds_load_b128s.
// =====================================================================
__global__ __launch_bounds__(256) void lstm_k(
    const float* __restrict__ x,      // [BN, T, 2]
    const float* __restrict__ W_ih,   // [256, 2]
    const float* __restrict__ W_hh,   // [256, 64]
    const float* __restrict__ b_ih,   // [256]
    const float* __restrict__ b_hh,   // [256]
    float* __restrict__ feats)        // [BN, 64]
{
  // Fragment-linear W_hh: sWp[tile][kt][lane][j] = W_hh[tile*16+(lane&15)]
  //                                                    [kt*32 + AK(j, lane>>4)]
  __shared__ _Float16 sWp[16][2][32][16];   // 32 KB
  // Fragment-linear h:   hA[wave][kt][lane][j]
  __shared__ _Float16 hA[8][2][32][16];     // 16 KB
  __shared__ float    sX[8][16][2];         // per-wave x_t

  const int tid = threadIdx.x;
  for (int i = tid; i < 16 * 2 * 32 * 16; i += 256) {
    int j = i & 15, L = (i >> 4) & 31, kt = (i >> 9) & 1, tile = i >> 10;
    int row = tile * 16 + (L & 15);
    int col = kt * 32 + AK(j, L >> 4);
    sWp[tile][kt][L][j] = (_Float16)W_hh[row * 64 + col];
  }
  for (int i = tid; i < 8 * 2 * 32 * 16; i += 256)
    ((_Float16*)hA)[i] = (_Float16)0.0f;
  __syncthreads();

  const int wv = tid >> 5, ln = tid & 31;
  const int half = ln >> 4, l15 = ln & 15;
  const int seq0 = (blockIdx.x * 8 + wv) * 16;

  // Hoist loop-invariant xp parameters into registers (per gate tile).
  float wi0[16], wi1[16], bs[16];
  #pragma unroll
  for (int tile = 0; tile < 16; ++tile) {
    int gidx = tile * 16 + l15;
    wi0[tile] = W_ih[gidx * 2 + 0];
    wi1[tile] = W_ih[gidx * 2 + 1];
    bs[tile]  = b_ih[gidx] + b_hh[gidx];
  }

  // Per-lane write coordinates for the fragment-linear h buffer:
  // for hidden tile k, this lane writes col = k*16+l15 at rows r+8*half.
  int wkt[4], wj[4], whr[4];
  #pragma unroll
  for (int k = 0; k < 4; ++k) {
    int col = k * 16 + l15;
    wkt[k] = col >> 5;
    int K = col & 31;
    whr[k] = (K >> 3) & 1;                 // reader half-wave
    wj[k]  = (K & 7) + ((K & 16) ? 8 : 0); // reader fragment index
  }

  v8f cst[4];
  #pragma unroll
  for (int k = 0; k < 4; ++k) cst[k] = v8f{};

  for (int t = 0; t < Tt; ++t) {
    if (ln < 16) {
      const float* xp = x + ((size_t)(seq0 + ln) * Tt + t) * 2;
      sX[wv][ln][0] = xp[0];
      sX[wv][ln][1] = xp[1];
      __builtin_prefetch(xp + 8, 0, 3);   // global_prefetch: x_{t+4}
    }
    lds_fence();  // make x_t and last step's h visible to all lanes of wave

    // h -> A fragments: contiguous v16h loads (2 x ds_load_b128 each)
    v16h a0 = *(const v16h*)(&hA[wv][0][ln][0]);
    v16h a1 = *(const v16h*)(&hA[wv][1][ln][0]);

    float xr0[8], xr1[8];
    #pragma unroll
    for (int r = 0; r < 8; ++r) {
      int row = r + 8 * half;
      xr0[r] = sX[wv][row][0];
      xr1[r] = sX[wv][row][1];
    }

    const bool last = (t == Tt - 1);
    #pragma unroll
    for (int k = 0; k < 4; ++k) {         // hidden tile (16 units)
      v8f g[4];
      #pragma unroll
      for (int gi = 0; gi < 4; ++gi) {    // i, f, g, o
        const int tile = gi * 4 + k;
        v8f acc;
        #pragma unroll
        for (int r = 0; r < 8; ++r)
          acc[r] = xr0[r] * wi0[tile] + xr1[r] * wi1[tile] + bs[tile];
        v16h b0v = *(const v16h*)(&sWp[tile][0][ln][0]);
        v16h b1v = *(const v16h*)(&sWp[tile][1][ln][0]);
        acc = wmma16(a0, b0v, acc);
        acc = wmma16(a1, b1v, acc);
        g[gi] = acc;
      }
      #pragma unroll
      for (int r = 0; r < 8; ++r) {
        float ig = sigmoid_(g[0][r]);
        float fg = sigmoid_(g[1][r]);
        float gg = tanh_(g[2][r]);
        float og = sigmoid_(g[3][r]);
        float cc = fg * cst[k][r] + ig * gg;
        cst[k][r] = cc;
        float hh = og * tanh_(cc);
        int row = r + 8 * half;
        if (last) feats[(size_t)(seq0 + row) * 64 + k * 16 + l15] = hh;
        else      hA[wv][wkt[k]][whr[k] * 16 + row][wj[k]] = (_Float16)hh;
      }
    }
  }
}

// =====================================================================
// Kernel 2: GCN. One scene per block. Adjacency fragments built on the
// fly from positions (no NxN materialization), then agg @ W_gcn + ReLU.
// =====================================================================
__global__ __launch_bounds__(256) void gcn_k(
    const float* __restrict__ pos,     // [B, N, 2]
    const float* __restrict__ feats,   // [BN, 64]
    const float* __restrict__ W_gcn,   // [64, 64]
    const float* __restrict__ b_gcn,   // [64]
    _Float16* __restrict__ gcnout)     // [BN, 64] f16
{
  __shared__ float    sPx[256], sPy[256], sDinv[256];
  __shared__ _Float16 sF[256][66];
  __shared__ _Float16 sWg[64][66];

  const int tid = threadIdx.x, b = blockIdx.x;
  sPx[tid] = pos[((size_t)b * 256 + tid) * 2 + 0];
  sPy[tid] = pos[((size_t)b * 256 + tid) * 2 + 1];
  #pragma unroll 4
  for (int k = 0; k < 64; ++k)
    sF[tid][k] = (_Float16)feats[((size_t)b * 256 + tid) * 64 + k];
  if (tid < 64)
    for (int e = 0; e < 64; ++e) sWg[tid][e] = (_Float16)W_gcn[tid * 64 + e];
  __syncthreads();

  int deg = 0;
  {
    float px = sPx[tid], py = sPy[tid];
    for (int j = 0; j < 256; ++j) {
      float dx = px - sPx[j], dy = py - sPy[j];
      if (dx * dx + dy * dy < THR2) deg++;
    }
  }
  sDinv[tid] = (deg > 0) ? rsqrtf((float)deg) : 0.0f;
  __syncthreads();

  const int wv = tid >> 5, ln = tid & 31;
  const int half = ln >> 4, l15 = ln & 15;

  v8f gout[2][4];
  #pragma unroll
  for (int mt = 0; mt < 2; ++mt) {
    const int Mbase = (wv * 2 + mt) * 16;
    const int irow = Mbase + l15;
    const float di = sDinv[irow], pix = sPx[irow], piy = sPy[irow];
    v8f acc[4];
    #pragma unroll
    for (int n = 0; n < 4; ++n) acc[n] = v8f{};
    for (int ch = 0; ch < 8; ++ch) {      // K chunks of 32 neighbors
      v16h a;
      #pragma unroll
      for (int j = 0; j < 16; ++j) {
        int jj = ch * 32 + AK(j, half);
        float dx = pix - sPx[jj], dy = piy - sPy[jj];
        float w = (dx * dx + dy * dy < THR2) ? di * sDinv[jj] : 0.0f;
        a[j] = (_Float16)w;
      }
      #pragma unroll
      for (int n = 0; n < 4; ++n) {
        v16h bv;
        #pragma unroll
        for (int j = 0; j < 16; ++j)
          bv[j] = sF[ch * 32 + AK(j, half)][n * 16 + l15];
        acc[n] = wmma16(a, bv, acc[n]);
      }
    }
    #pragma unroll
    for (int n = 0; n < 4; ++n) gout[mt][n] = acc[n];
  }
  __syncthreads();                       // everyone done reading sF
  #pragma unroll
  for (int mt = 0; mt < 2; ++mt) {
    const int Mbase = (wv * 2 + mt) * 16;
    #pragma unroll
    for (int n = 0; n < 4; ++n)
      #pragma unroll
      for (int r = 0; r < 8; ++r)
        sF[Mbase + r + 8 * half][n * 16 + l15] = (_Float16)gout[mt][n][r];
  }
  __syncthreads();

  #pragma unroll
  for (int mt = 0; mt < 2; ++mt) {
    const int Mbase = (wv * 2 + mt) * 16;
    v16h a0, a1;
    #pragma unroll
    for (int j = 0; j < 16; ++j) {
      int k = AK(j, half);
      a0[j] = sF[Mbase + l15][k];
      a1[j] = sF[Mbase + l15][32 + k];
    }
    #pragma unroll
    for (int n = 0; n < 4; ++n) {
      float bb = b_gcn[n * 16 + l15];
      v8f acc;
      #pragma unroll
      for (int r = 0; r < 8; ++r) acc[r] = bb;
      v16h b0, b1;
      #pragma unroll
      for (int j = 0; j < 16; ++j) {
        int k = AK(j, half);
        b0[j] = sWg[k][n * 16 + l15];     // B(k,e) = W_gcn[k][e]
        b1[j] = sWg[32 + k][n * 16 + l15];
      }
      acc = wmma16(a0, b0, acc);
      acc = wmma16(a1, b1, acc);
      #pragma unroll
      for (int r = 0; r < 8; ++r) {
        float v = fmaxf(acc[r], 0.0f);
        gcnout[((size_t)b * 256 + Mbase + r + 8 * half) * 64 + n * 16 + l15] =
            (_Float16)v;
      }
    }
  }
}

// =====================================================================
// Kernel 3a: packed QKV projection (torch: y = x @ W_in^T + b_in)
// =====================================================================
__global__ __launch_bounds__(256) void qkv_k(
    const _Float16* __restrict__ gcn,  // [BN, 64] f16
    const float* __restrict__ W_in,    // [192, 64]
    const float* __restrict__ b_in,    // [192]
    _Float16* __restrict__ qkv)        // [BN, 192] f16
{
  __shared__ _Float16 sG[256][66];
  __shared__ _Float16 sWin[192][66];
  __shared__ float    sBin[192];

  const int tid = threadIdx.x, b = blockIdx.x;
  #pragma unroll 4
  for (int k = 0; k < 64; ++k)
    sG[tid][k] = gcn[((size_t)b * 256 + tid) * 64 + k];
  if (tid < 192) {
    for (int k = 0; k < 64; ++k) sWin[tid][k] = (_Float16)W_in[tid * 64 + k];
    sBin[tid] = b_in[tid];
  }
  __syncthreads();

  const int wv = tid >> 5, ln = tid & 31;
  const int half = ln >> 4, l15 = ln & 15;

  #pragma unroll
  for (int mt = 0; mt < 2; ++mt) {
    const int Mbase = (wv * 2 + mt) * 16;
    v16h a0, a1;
    #pragma unroll
    for (int j = 0; j < 16; ++j) {
      int k = AK(j, half);
      a0[j] = sG[Mbase + l15][k];
      a1[j] = sG[Mbase + l15][32 + k];
    }
    for (int n = 0; n < 12; ++n) {
      float bb = sBin[n * 16 + l15];
      v8f acc;
      #pragma unroll
      for (int r = 0; r < 8; ++r) acc[r] = bb;
      v16h b0, b1;
      #pragma unroll
      for (int j = 0; j < 16; ++j) {
        int k = AK(j, half);
        b0[j] = sWin[n * 16 + l15][k];    // B(k,n) = W_in[n][k]
        b1[j] = sWin[n * 16 + l15][32 + k];
      }
      acc = wmma16(a0, b0, acc);
      acc = wmma16(a1, b1, acc);
      #pragma unroll
      for (int r = 0; r < 8; ++r)
        qkv[((size_t)b * 256 + Mbase + r + 8 * half) * 192 + n * 16 + l15] =
            (_Float16)acc[r];
    }
  }
}

// =====================================================================
// Kernel 3b: one (scene, head) per block. head_dim=16 padded to K=32.
// Softmax via intra-half-wave shfl_xor reductions (wave32 layout).
// =====================================================================
__global__ __launch_bounds__(256) void attn_k(
    const _Float16* __restrict__ qkv,  // [BN, 192] f16
    _Float16* __restrict__ obuf)       // [BN, 64] f16
{
  __shared__ _Float16 sQ[256][18], sK[256][18], sV[256][18];
  __shared__ _Float16 sP[8][16][34];   // per-wave prob chunk (16x32)

  const int tid = threadIdx.x;
  const int scene = blockIdx.x >> 2, head = blockIdx.x & 3;
  const size_t base = ((size_t)scene * 256 + tid) * 192 + head * 16;
  #pragma unroll
  for (int d = 0; d < 16; ++d) {
    sQ[tid][d] = (_Float16)((float)qkv[base + d] * 0.25f);  // 1/sqrt(16)
    sK[tid][d] = qkv[base + 64 + d];
    sV[tid][d] = qkv[base + 128 + d];
  }
  __syncthreads();

  const int wv = tid >> 5, ln = tid & 31;
  const int half = ln >> 4, l15 = ln & 15;

  #pragma unroll
  for (int mt = 0; mt < 2; ++mt) {
    const int Mbase = (wv * 2 + mt) * 16;
    v16h a;
    #pragma unroll
    for (int j = 0; j < 16; ++j) {
      int k = AK(j, half);
      a[j] = (k < 16) ? sQ[Mbase + l15][k] : (_Float16)0.0f;
    }
    v8f s[16];
    for (int n = 0; n < 16; ++n) {
      v16h bv;
      #pragma unroll
      for (int j = 0; j < 16; ++j) {
        int k = AK(j, half);
        bv[j] = (k < 16) ? sK[n * 16 + l15][k] : (_Float16)0.0f;
      }
      v8f z = {};
      s[n] = wmma16(a, bv, z);
    }
    // row softmax: rows live in half-waves of 16 lanes
    float inv[8];
    #pragma unroll
    for (int r = 0; r < 8; ++r) {
      float m = -3.0e30f;
      #pragma unroll
      for (int n = 0; n < 16; ++n) m = fmaxf(m, s[n][r]);
      #pragma unroll
      for (int off = 8; off >= 1; off >>= 1) m = fmaxf(m, __shfl_xor(m, off, 32));
      float sum = 0.0f;
      #pragma unroll
      for (int n = 0; n < 16; ++n) { float e = __expf(s[n][r] - m); s[n][r] = e; sum += e; }
      #pragma unroll
      for (int off = 8; off >= 1; off >>= 1) sum += __shfl_xor(sum, off, 32);
      inv[r] = 1.0f / sum;
    }
    // attn @ V, K in chunks of 32 keys
    v8f oa = {};
    for (int ch = 0; ch < 8; ++ch) {
      #pragma unroll
      for (int nn = 0; nn < 2; ++nn) {
        int n = ch * 2 + nn;
        #pragma unroll
        for (int r = 0; r < 8; ++r)
          sP[wv][r + 8 * half][nn * 16 + l15] = (_Float16)(s[n][r] * inv[r]);
      }
      lds_fence();
      v16h ap, bv;
      #pragma unroll
      for (int j = 0; j < 16; ++j) {
        int k = AK(j, half);
        ap[j] = sP[wv][l15][k];
        bv[j] = sV[ch * 32 + k][l15];     // B(k,d) = V[key k][d]
      }
      oa = wmma16(ap, bv, oa);
    }
    #pragma unroll
    for (int r = 0; r < 8; ++r)
      obuf[((size_t)scene * 256 + Mbase + r + 8 * half) * 64 + head * 16 + l15] =
          (_Float16)oa[r];
  }
}

// =====================================================================
// Kernel 3c: out-proj (x @ W_out^T + b) -> ctrl (x @ W_dec + b) -> Bezier
// =====================================================================
__global__ __launch_bounds__(256) void dec_k(
    const _Float16* __restrict__ obuf, // [BN, 64] f16
    const float* __restrict__ W_out,   // [64, 64]
    const float* __restrict__ b_out,   // [64]
    const float* __restrict__ W_dec,   // [64, 8]
    const float* __restrict__ b_dec,   // [8]
    float* __restrict__ out)           // [B, N, FUT, 2]
{
  __shared__ _Float16 sO[256][66];
  __shared__ _Float16 sWo[64][66];
  __shared__ _Float16 sWd[64][18];
  __shared__ float    sCtrl[256][8];

  const int tid = threadIdx.x, b = blockIdx.x;
  #pragma unroll 4
  for (int k = 0; k < 64; ++k)
    sO[tid][k] = obuf[((size_t)b * 256 + tid) * 64 + k];
  if (tid < 64) {
    for (int e = 0; e < 64; ++e) sWo[tid][e] = (_Float16)W_out[tid * 64 + e];
    for (int e = 0; e < 8; ++e)  sWd[tid][e] = (_Float16)W_dec[tid * 8 + e];
  }
  __syncthreads();

  const int wv = tid >> 5, ln = tid & 31;
  const int half = ln >> 4, l15 = ln & 15;

  #pragma unroll
  for (int mt = 0; mt < 2; ++mt) {
    const int Mbase = (wv * 2 + mt) * 16;
    v16h a0, a1;
    #pragma unroll
    for (int j = 0; j < 16; ++j) {
      int k = AK(j, half);
      a0[j] = sO[Mbase + l15][k];
      a1[j] = sO[Mbase + l15][32 + k];
    }
    v8f y[4];
    #pragma unroll
    for (int n = 0; n < 4; ++n) {
      float bb = b_out[n * 16 + l15];
      v8f acc;
      #pragma unroll
      for (int r = 0; r < 8; ++r) acc[r] = bb;
      v16h b0, b1;
      #pragma unroll
      for (int j = 0; j < 16; ++j) {
        int k = AK(j, half);
        b0[j] = sWo[n * 16 + l15][k];     // B(k,n) = W_out[n][k]
        b1[j] = sWo[n * 16 + l15][32 + k];
      }
      acc = wmma16(a0, b0, acc);
      acc = wmma16(a1, b1, acc);
      y[n] = acc;
    }
    // stash y back into this wave's private rows of sO (f16) for next GEMM
    #pragma unroll
    for (int n = 0; n < 4; ++n)
      #pragma unroll
      for (int r = 0; r < 8; ++r)
        sO[Mbase + r + 8 * half][n * 16 + l15] = (_Float16)y[n][r];
    lds_fence();

    v16h c0, c1;
    #pragma unroll
    for (int j = 0; j < 16; ++j) {
      int k = AK(j, half);
      c0[j] = sO[Mbase + l15][k];
      c1[j] = sO[Mbase + l15][32 + k];
    }
    v8f cacc;
    #pragma unroll
    for (int r = 0; r < 8; ++r) cacc[r] = (l15 < 8) ? b_dec[l15] : 0.0f;
    v16h d0, d1;
    #pragma unroll
    for (int j = 0; j < 16; ++j) {
      int k = AK(j, half);
      d0[j] = (l15 < 8) ? sWd[k][l15] : (_Float16)0.0f;       // B(k,c) = W_dec[k][c]
      d1[j] = (l15 < 8) ? sWd[32 + k][l15] : (_Float16)0.0f;
    }
    cacc = wmma16(c0, d0, cacc);
    cacc = wmma16(c1, d1, cacc);
    #pragma unroll
    for (int r = 0; r < 8; ++r)
      if (l15 < 8) sCtrl[Mbase + r + 8 * half][l15] = cacc[r];
  }
  __syncthreads();

  // Bezier basis (degree 3): coeffs {1,3,3,1}
  for (int idx = tid; idx < 256 * FUT * 2; idx += 256) {
    int n = idx / (FUT * 2);
    int rem = idx % (FUT * 2);
    int t = rem >> 1, c = rem & 1;
    float tt = (float)t * (1.0f / (float)(FUT - 1));
    float u = 1.0f - tt;
    float w0 = u * u * u;
    float w1 = 3.0f * tt * u * u;
    float w2 = 3.0f * tt * tt * u;
    float w3 = tt * tt * tt;
    float v = w0 * sCtrl[n][0 + c] + w1 * sCtrl[n][2 + c] +
              w2 * sCtrl[n][4 + c] + w3 * sCtrl[n][6 + c];
    out[((size_t)b * 256 + n) * (FUT * 2) + rem] = v;
  }
}

// =====================================================================
extern "C" void kernel_launch(void* const* d_in, const int* in_sizes, int n_in,
                              void* d_out, int out_size, void* d_ws, size_t ws_size,
                              hipStream_t stream) {
  const float* x     = (const float*)d_in[0];   // agent_histories
  const float* pos   = (const float*)d_in[1];   // agent_positions
  const float* W_ih  = (const float*)d_in[2];
  const float* W_hh  = (const float*)d_in[3];
  const float* b_ih  = (const float*)d_in[4];
  const float* b_hh  = (const float*)d_in[5];
  const float* W_gcn = (const float*)d_in[6];
  const float* b_gcn = (const float*)d_in[7];
  const float* W_in  = (const float*)d_in[8];
  const float* b_in  = (const float*)d_in[9];
  const float* W_out = (const float*)d_in[10];
  const float* b_out = (const float*)d_in[11];
  const float* W_dec = (const float*)d_in[12];
  const float* b_dec = (const float*)d_in[13];
  float* out = (float*)d_out;

  char* ws = (char*)d_ws;
  float*     feats = (float*)    (ws);                        //  8 MiB f32 [BN,64]
  _Float16*  gcnb  = (_Float16*) (ws +  8u * 1024 * 1024);    //  4 MiB f16 [BN,64]
  _Float16*  qkvb  = (_Float16*) (ws + 12u * 1024 * 1024);    // 12 MiB f16 [BN,192]
  _Float16*  obuf  = (_Float16*) (ws + 24u * 1024 * 1024);    //  4 MiB f16 [BN,64]

  lstm_k<<<BN / (16 * 8), 256, 0, stream>>>(x, W_ih, W_hh, b_ih, b_hh, feats);
  gcn_k <<<Bc, 256, 0, stream>>>(pos, feats, W_gcn, b_gcn, gcnb);
  qkv_k <<<Bc, 256, 0, stream>>>(gcnb, W_in, b_in, qkvb);
  attn_k<<<Bc * 4, 256, 0, stream>>>(qkvb, obuf);
  dec_k <<<Bc, 256, 0, stream>>>(obuf, W_out, b_out, W_dec, b_dec, out);
}
